// MultiHeadedAttention_66331474919935
// MI455X (gfx1250) — matrix operand order repaired and checked by reading, not compiled, not run on hardware
//
#include <hip/hip_runtime.h>
#include <hip/hip_bf16.h>
#include <math.h>

// MHA forward: B=4, L=2048, D=1024, H=16, DK=64 on gfx1250 (MI455X).
// Pipeline:
//   k0: weights fp32 -> f16 (once; weights stay L2-resident, 16 MB total traffic)
//   k1: QKV projection, f16 WMMA / fp32 acc; V stored transposed [B,H,DK,L]
//   k2: causal flash attention, f16 WMMA, fp32 online softmax
//   k3: output projection, f16 WMMA, fp32 out + bias
// Tile staging of f16 operands uses GLOBAL_LOAD_ASYNC_TO_LDS_B128 (ASYNCcnt).
// Workspace layout (needs 72 MB):
//   [ 0,16) MB  Qh   f16 [B,H,L,DK]
//   [16,32) MB  Kh   f16 [B,H,L,DK]
//   [32,48) MB  Vt   f16 [B,H,DK,L]   (transposed for PV B-fragments)
//   [48,64) MB  Ch   f16 [B*L, D]     (attention context)
//   [64,72) MB  Wh   f16 Wq|Wk|Wv|Wo, 1M elements each

typedef __attribute__((ext_vector_type(16))) _Float16 v16h;
typedef __attribute__((ext_vector_type(8)))  _Float16 v8h;
typedef __attribute__((ext_vector_type(4)))  _Float16 v4h;
typedef __attribute__((ext_vector_type(8)))  float    v8f;

#define WMMA_F16(a, b, c) \
  __builtin_amdgcn_wmma_f32_16x16x32_f16(false, (a), false, (b), (short)0, (c), false, false)

// Async DMA: copy 16B from global (SADDR base + 32-bit voffset) into LDS at
// lds_addr (low 32 bits of flat LDS pointer).  Tracked by ASYNCcnt.
__device__ __forceinline__ void async_b128(uint32_t lds_addr, const void* base,
                                           uint32_t voff_bytes) {
  asm volatile("global_load_async_to_lds_b128 %0, %1, %2"
               :
               : "v"(lds_addr), "v"(voff_bytes), "s"(base)
               : "memory");
}
__device__ __forceinline__ void wait_async0() {
  asm volatile("s_wait_asynccnt 0" ::: "memory");
}
__device__ __forceinline__ uint32_t lds_addr_of(const void* p) {
  return (uint32_t)(uintptr_t)p;  // LDS aperture: addr[31:0] is the LDS offset
}

// ---- WMMA fragment builders (ISA 7.12.2 layouts, wave32) ----------------
// A (16x32 f16): lane l holds row m=l&15, kb=(l<16)?0:8;
// elems 0..7 = K[kb..kb+7], elems 8..15 = K[16+kb..16+kb+7].
template <int STRIDE>
__device__ __forceinline__ v16h a_frag(const _Float16 (*tile)[STRIDE], int row0,
                                       int kcol0, int lane) {
  const int m  = row0 + (lane & 15);
  const int kb = (lane & 16) ? 8 : 0;
  union { v16h v; v8h x[2]; } u;
  u.x[0] = *reinterpret_cast<const v8h*>(&tile[m][kcol0 + kb]);
  u.x[1] = *reinterpret_cast<const v8h*>(&tile[m][kcol0 + kb + 16]);
  return u.v;
}
// B (32x16 f16): lane l holds col n=l&15, kb=(l<16)?0:8;
// elems 0..7 = K[kb..kb+7], elems 8..15 = K[16+kb..16+kb+7].
// Tile stored [n][k] so the 16 K values are contiguous.
template <int STRIDE>
__device__ __forceinline__ v16h b_frag(const _Float16 (*tile)[STRIDE], int row0,
                                       int kcol0, int lane) {
  const int n  = row0 + (lane & 15);
  const int kb = (lane & 16) ? 8 : 0;
  union { v16h v; v8h x[2]; } u;
  u.x[0] = *reinterpret_cast<const v8h*>(&tile[n][kcol0 + kb]);
  u.x[1] = *reinterpret_cast<const v8h*>(&tile[n][kcol0 + kb + 16]);
  return u.v;
}

// ---- kernel 0: one-shot weight convert fp32 -> f16 ----------------------
// grid (1024, 4), block 256; 4 floats/thread.
__global__ __launch_bounds__(256) void mha_cvt_weights(
    const float* __restrict__ Wq, const float* __restrict__ Wk,
    const float* __restrict__ Wv, const float* __restrict__ Wo,
    _Float16* __restrict__ dst) {
  const int z = blockIdx.y;
  const float* src = (z == 0) ? Wq : (z == 1) ? Wk : (z == 2) ? Wv : Wo;
  _Float16* d = dst + (size_t)z * (1024 * 1024);
  const int i = (blockIdx.x * 256 + threadIdx.x) * 4;
  const float4 f = *reinterpret_cast<const float4*>(src + i);
  v4h h = {(_Float16)f.x, (_Float16)f.y, (_Float16)f.z, (_Float16)f.w};
  *reinterpret_cast<v4h*>(d + i) = h;
}

// ---- kernel 1: fused QKV projection ------------------------------------
// grid (16, 128, 3), block 128.  Y[m,n] = sum_d X[m,d]*Wh[n,d] + b[n].
// z==2 (V) is stored transposed [B,H,DK,L]; else [B,H,L,DK].
__global__ __launch_bounds__(128) void mha_proj_qkv(
    const float* __restrict__ Xq, const float* __restrict__ Xk,
    const float* __restrict__ Xv, const _Float16* __restrict__ Wh4,
    const float* __restrict__ bq, const float* __restrict__ bk,
    const float* __restrict__ bv,
    _Float16* __restrict__ Qh, _Float16* __restrict__ Kh,
    _Float16* __restrict__ Vt) {
  constexpr int KD = 1024;
  const int z = blockIdx.z;
  const float* X       = (z == 0) ? Xq : (z == 1) ? Xk : Xv;
  const _Float16* Wh   = Wh4 + (size_t)z * (1024 * 1024);
  const float* bs      = (z == 0) ? bq : (z == 1) ? bk : bv;
  _Float16*   dst      = (z == 0) ? Qh : (z == 1) ? Kh : Vt;

  __shared__ alignas(16) _Float16 At[64][40];  // [m][k], 80B rows (16B mult)
  __shared__ alignas(16) _Float16 Bt[64][40];  // [n][k] = W rows

  const int tid = threadIdx.x;
  const int lane = tid & 31;
  const int wv = tid >> 5;
  const int wm = (wv >> 1) * 32;
  const int wn = (wv & 1) * 32;
  const int m0 = blockIdx.y * 64;
  const int n0 = blockIdx.x * 64;

  v8f acc[2][2] = {};

  for (int k0 = 0; k0 < KD; k0 += 32) {
    __syncthreads();
    // A tile: fp32 activations -> f16 in LDS (must convert, VALU path)
#pragma unroll
    for (int p = 0; p < 4; ++p) {
      const int idx = p * 512 + tid * 4;
      const int r = idx >> 5, c = idx & 31;
      const float4 fa =
          *reinterpret_cast<const float4*>(X + (size_t)(m0 + r) * KD + k0 + c);
      v4h ha = {(_Float16)fa.x, (_Float16)fa.y, (_Float16)fa.z, (_Float16)fa.w};
      *reinterpret_cast<v4h*>(&At[r][c]) = ha;
    }
    // B tile: pre-converted f16 weights, async DMA into LDS
#pragma unroll
    for (int p = 0; p < 2; ++p) {
      const int linear = p * 128 + tid;     // 64 rows x 4 chunks of 8 halfs
      const int r = linear >> 2, c = (linear & 3) * 8;
      async_b128(lds_addr_of(&Bt[r][c]), Wh,
                 (uint32_t)(((n0 + r) * KD + k0 + c) * 2));
    }
    wait_async0();
    __syncthreads();
    const v16h af0 = a_frag<40>(At, wm, 0, lane);
    const v16h af1 = a_frag<40>(At, wm + 16, 0, lane);
    const v16h bf0 = b_frag<40>(Bt, wn, 0, lane);
    const v16h bf1 = b_frag<40>(Bt, wn + 16, 0, lane);
    acc[0][0] = WMMA_F16(af0, bf0, acc[0][0]);
    acc[0][1] = WMMA_F16(af0, bf1, acc[0][1]);
    acc[1][0] = WMMA_F16(af1, bf0, acc[1][0]);
    acc[1][1] = WMMA_F16(af1, bf1, acc[1][1]);
  }

  const int hi = (lane >> 4) & 1, cl = lane & 15;
#pragma unroll
  for (int i = 0; i < 2; ++i)
#pragma unroll
    for (int j = 0; j < 2; ++j)
#pragma unroll
      for (int r = 0; r < 8; ++r) {
        const int row = m0 + wm + i * 16 + r + 8 * hi;  // b*L + l
        const int col = n0 + wn + j * 16 + cl;          // h*64 + dk
        const float val = acc[i][j][r] + bs[col];
        const size_t bhp = (size_t)(row >> 11) * 16 + (col >> 6);  // b*H + h
        size_t o;
        if (z == 2)  // V transposed: [b,h,dk,l]
          o = (bhp * 64 + (col & 63)) * 2048 + (row & 2047);
        else         // [b,h,l,dk]
          o = (bhp * 2048 + (row & 2047)) * 64 + (col & 63);
        dst[o] = (_Float16)val;
      }
}

// ---- kernel 2: causal flash attention ----------------------------------
// grid (32, 64), block 128 (4 waves; wave w owns q rows w*16..w*16+15).
__global__ __launch_bounds__(128) void mha_flash_attn(
    const _Float16* __restrict__ Qh, const _Float16* __restrict__ Kh,
    const _Float16* __restrict__ Vth, _Float16* __restrict__ ctx) {
  constexpr int L = 2048, DK = 64;
  const int qt = blockIdx.x;
  const int bh = blockIdx.y;
  const int b = bh >> 4, h = bh & 15;
  const _Float16* Q = Qh  + (size_t)bh * L * DK;   // [l][dk]
  const _Float16* K = Kh  + (size_t)bh * L * DK;   // [l][dk]
  const _Float16* V = Vth + (size_t)bh * DK * L;   // [dk][l]  (transposed)

  __shared__ alignas(16) _Float16 Kt[64][72];     // [key][dk]: B-frag for QK^T
  __shared__ alignas(16) _Float16 Vt[64][72];     // [dk][key]: B-frag for PV
  __shared__ alignas(16) _Float16 Pw[4][16][72];  // per-wave P (A-frag source)

  const int tid = threadIdx.x, lane = tid & 31, wv = tid >> 5;
  const int hi = (lane >> 4) & 1, cl = lane & 15;

  // Q fragments (reused across all key tiles) straight from global.
  v16h qf[2];
  {
    const _Float16* qrow = Q + (size_t)(qt * 64 + wv * 16 + cl) * DK;
    const int kb = hi ? 8 : 0;
#pragma unroll
    for (int s = 0; s < 2; ++s) {
      union { v16h v; v8h x[2]; } u;
      u.x[0] = *reinterpret_cast<const v8h*>(qrow + s * 32 + kb);
      u.x[1] = *reinterpret_cast<const v8h*>(qrow + s * 32 + kb + 16);
      qf[s] = u.v;
    }
  }

  v8f o[4] = {};
  float mrow[8], lrow[8];
#pragma unroll
  for (int r = 0; r < 8; ++r) { mrow[r] = -3.0e38f; lrow[r] = 0.0f; }
  const float scale = 0.125f;  // 1/sqrt(64)

  for (int kt = 0; kt <= qt; ++kt) {
    __syncthreads();
    // Async DMA staging: both tiles are contiguous 128B row copies now.
#pragma unroll
    for (int p = 0; p < 4; ++p) {
      const int linear = p * 128 + tid;   // 64 rows x 8 chunks of 8 halfs
      const int r = linear >> 3, c = (linear & 7) * 8;
      async_b128(lds_addr_of(&Kt[r][c]), K,
                 (uint32_t)(((kt * 64 + r) * DK + c) * 2));
      async_b128(lds_addr_of(&Vt[r][c]), V,
                 (uint32_t)((r * L + kt * 64 + c) * 2));
    }
    if (kt < qt) {  // prefetch next tiles into cache while this one computes
      __builtin_prefetch(K + ((size_t)(kt + 1) * 64 + (tid >> 1)) * DK +
                             (tid & 1) * 32, 0, 3);
      __builtin_prefetch(V + (size_t)(tid >> 1) * L + (kt + 1) * 64 +
                             (tid & 1) * 32, 0, 3);
    }
    wait_async0();
    __syncthreads();

    // S = Q * K^T  (16 q rows x 64 keys per wave)
    v8f s4[4] = {};
#pragma unroll
    for (int ks = 0; ks < 2; ++ks)
#pragma unroll
      for (int t = 0; t < 4; ++t) {
        const v16h bf = b_frag<72>(Kt, t * 16, ks * 32, lane);
        s4[t] = WMMA_F16(qf[ks], bf, s4[t]);
      }

    const bool diag = (kt == qt);
#pragma unroll
    for (int t = 0; t < 4; ++t)
#pragma unroll
      for (int r = 0; r < 8; ++r) {
        float sv = s4[t][r] * scale;
        if (diag) {
          const int rowg = wv * 16 + r + 8 * hi;
          const int colg = t * 16 + cl;
          if (colg > rowg) sv = -3.0e38f;
        }
        s4[t][r] = sv;
      }

    // Online softmax; each C row lives in a 16-lane half-group.
#pragma unroll
    for (int r = 0; r < 8; ++r) {
      float mx = fmaxf(fmaxf(s4[0][r], s4[1][r]), fmaxf(s4[2][r], s4[3][r]));
#pragma unroll
      for (int msk = 1; msk < 16; msk <<= 1)
        mx = fmaxf(mx, __shfl_xor(mx, msk, 32));
      const float mnew = fmaxf(mrow[r], mx);
      const float alpha = __expf(mrow[r] - mnew);
      mrow[r] = mnew;
      float rs = 0.0f;
#pragma unroll
      for (int t = 0; t < 4; ++t) {
        const float pv = __expf(s4[t][r] - mnew);
        s4[t][r] = pv;
        rs += pv;
      }
#pragma unroll
      for (int msk = 1; msk < 16; msk <<= 1) rs += __shfl_xor(rs, msk, 32);
      lrow[r] = lrow[r] * alpha + rs;
#pragma unroll
      for (int t = 0; t < 4; ++t) o[t][r] *= alpha;
    }

    // P: C-layout -> per-wave LDS -> A-layout
#pragma unroll
    for (int t = 0; t < 4; ++t)
#pragma unroll
      for (int r = 0; r < 8; ++r)
        Pw[wv][r + 8 * hi][t * 16 + cl] = (_Float16)s4[t][r];

    // O += P * V
#pragma unroll
    for (int ks = 0; ks < 2; ++ks) {
      const v16h pf = a_frag<72>(Pw[wv], 0, ks * 32, lane);
#pragma unroll
      for (int t = 0; t < 4; ++t) {
        const v16h vf = b_frag<72>(Vt, t * 16, ks * 32, lane);
        o[t] = WMMA_F16(pf, vf, o[t]);
      }
    }
  }

  // Normalize, emit ctx f16 in [B*L, D] layout (out-proj input).
#pragma unroll
  for (int t = 0; t < 4; ++t)
#pragma unroll
    for (int r = 0; r < 8; ++r) {
      const float val = o[t][r] / lrow[r];
      const int row = qt * 64 + wv * 16 + r + 8 * hi;
      const int col = h * 64 + t * 16 + cl;
      ctx[((size_t)b * L + row) * 1024 + col] = (_Float16)val;
    }
}

// ---- kernel 3: output projection (f16 in, fp32 out) --------------------
// grid (16, 128), block 128.  Both tiles async-staged (already f16).
__global__ __launch_bounds__(128) void mha_out_proj(
    const _Float16* __restrict__ Xh, const _Float16* __restrict__ Wh,
    const float* __restrict__ bs, float* __restrict__ Y) {
  constexpr int KD = 1024;
  __shared__ alignas(16) _Float16 At[64][40];
  __shared__ alignas(16) _Float16 Bt[64][40];

  const int tid = threadIdx.x;
  const int lane = tid & 31;
  const int wv = tid >> 5;
  const int wm = (wv >> 1) * 32;
  const int wn = (wv & 1) * 32;
  const int m0 = blockIdx.y * 64;
  const int n0 = blockIdx.x * 64;

  v8f acc[2][2] = {};

  for (int k0 = 0; k0 < KD; k0 += 32) {
    __syncthreads();
#pragma unroll
    for (int p = 0; p < 2; ++p) {
      const int linear = p * 128 + tid;
      const int r = linear >> 2, c = (linear & 3) * 8;
      async_b128(lds_addr_of(&At[r][c]), Xh,
                 (uint32_t)(((m0 + r) * KD + k0 + c) * 2));
      async_b128(lds_addr_of(&Bt[r][c]), Wh,
                 (uint32_t)(((n0 + r) * KD + k0 + c) * 2));
    }
    wait_async0();
    __syncthreads();
    const v16h af0 = a_frag<40>(At, wm, 0, lane);
    const v16h af1 = a_frag<40>(At, wm + 16, 0, lane);
    const v16h bf0 = b_frag<40>(Bt, wn, 0, lane);
    const v16h bf1 = b_frag<40>(Bt, wn + 16, 0, lane);
    acc[0][0] = WMMA_F16(af0, bf0, acc[0][0]);
    acc[0][1] = WMMA_F16(af0, bf1, acc[0][1]);
    acc[1][0] = WMMA_F16(af1, bf0, acc[1][0]);
    acc[1][1] = WMMA_F16(af1, bf1, acc[1][1]);
  }

  const int hi = (lane >> 4) & 1, cl = lane & 15;
#pragma unroll
  for (int i = 0; i < 2; ++i)
#pragma unroll
    for (int j = 0; j < 2; ++j)
#pragma unroll
      for (int r = 0; r < 8; ++r) {
        const int row = m0 + wm + i * 16 + r + 8 * hi;
        const int col = n0 + wn + j * 16 + cl;
        Y[(size_t)row * 1024 + col] = acc[i][j][r] + bs[col];
      }
}

extern "C" void kernel_launch(void* const* d_in, const int* in_sizes, int n_in,
                              void* d_out, int out_size, void* d_ws,
                              size_t ws_size, hipStream_t stream) {
  const float* query = (const float*)d_in[0];
  const float* key_  = (const float*)d_in[1];
  const float* value = (const float*)d_in[2];
  const float* Wq = (const float*)d_in[3];
  const float* bq = (const float*)d_in[4];
  const float* Wk = (const float*)d_in[5];
  const float* bk = (const float*)d_in[6];
  const float* Wv = (const float*)d_in[7];
  const float* bv = (const float*)d_in[8];
  const float* Wo = (const float*)d_in[9];
  const float* bo = (const float*)d_in[10];
  // d_in[11] is the causal bool mask; causality is hardcoded in the kernel.

  char* ws = (char*)d_ws;
  const size_t SZ = (size_t)8192 * 1024 * sizeof(_Float16);  // 16 MB
  _Float16* Qh = (_Float16*)(ws);
  _Float16* Kh = (_Float16*)(ws + SZ);
  _Float16* Vt = (_Float16*)(ws + 2 * SZ);
  _Float16* Ch = (_Float16*)(ws + 3 * SZ);
  _Float16* Wh = (_Float16*)(ws + 4 * SZ);  // 4 x 1M f16 = 8 MB
  const size_t WSZ = (size_t)1024 * 1024;

  dim3 blk(128);
  mha_cvt_weights<<<dim3(1024, 4), dim3(256), 0, stream>>>(Wq, Wk, Wv, Wo, Wh);
  mha_proj_qkv<<<dim3(16, 128, 3), blk, 0, stream>>>(
      query, key_, value, Wh, bq, bk, bv, Qh, Kh, Vt);
  mha_flash_attn<<<dim3(32, 64), blk, 0, stream>>>(Qh, Kh, Vt, Ch);
  mha_out_proj<<<dim3(16, 128), blk, 0, stream>>>(Ch, Wh + 3 * WSZ, bo,
                                                  (float*)d_out);
}
